// STGN_72103910965804
// MI455X (gfx1250) — matrix operand order repaired and checked by reading, not compiled, not run on hardware
//
#include <hip/hip_runtime.h>

// ---------------------------------------------------------------------------
// STGN (time/distance-gated LSTM)  B=256 S=256 H=512 IN=512
// Phase 1: big WMMA GEMM  X[65536x512] @ Wx_all[512x4096]  (bf16 -> f32 acc)
//          with double-buffered GLOBAL_LOAD_ASYNC_TO_LDS_B128 staging.
// Phase 2: 256 sequential per-step kernels, each fusing
//          h @ {Whi,Whf,Whc,Who} (WMMA) + pointwise cell update.
// ---------------------------------------------------------------------------

#define BB   256
#define SS   256
#define HH   512
#define ININ 512
#define MROWS (BB * SS)          // 65536
#define NX    (8 * HH)           // 4096  (xi xf xc pt1 pt2 pd1 pd2 po)
#define NR    (4 * HH)           // 2048  (i f c o recurrent)

typedef __attribute__((ext_vector_type(16))) __bf16 bf16x16;
typedef __attribute__((ext_vector_type(8)))  float  floatx8;
typedef unsigned short ushort_t;

union Frag {
    bf16x16 v;
    uint4   q[2];
};

__device__ __forceinline__ ushort_t f2bf(float f) {
    union { float f; unsigned u; } x; x.f = f;
    unsigned r = x.u + 0x7FFFu + ((x.u >> 16) & 1u);   // round-to-nearest-even
    return (ushort_t)(r >> 16);
}
__device__ __forceinline__ float sigf(float x) { return 1.0f / (1.0f + __expf(-x)); }

// CDNA5 async copy: global -> LDS, 16 bytes per lane, tracked by ASYNCcnt.
__device__ __forceinline__ void async_copy_b128(unsigned lds_byte_off, const void* gptr) {
    asm volatile("global_load_async_to_lds_b128 %0, %1, off"
                 :: "v"(lds_byte_off), "v"(gptr) : "memory");
}
__device__ __forceinline__ void wait_async0() {
    asm volatile("s_wait_asynccnt 0x0" ::: "memory");
}

// ---------------------------------------------------------------------------
// Pack kernels
// ---------------------------------------------------------------------------
__global__ void pack_x_kernel(const float* __restrict__ in, ushort_t* __restrict__ xbf,
                              float* __restrict__ tt, float* __restrict__ dt) {
    long idx = (long)blockIdx.x * 256 + threadIdx.x;        // over 65536*512
    int m = (int)(idx >> 9);
    int k = (int)(idx & 511);
    const float* row = in + (long)m * (ININ + 2);
    xbf[idx] = f2bf(row[2 + k]);
    if (k == 0) { tt[m] = row[0]; dt[m] = row[1]; }
}

struct WPtrs { const float* p[12]; };   // 0..7 Wx (i,f,c,t1,t2,d1,d2,o), 8..11 Wh (i,f,c,o)

__global__ void pack_w_kernel(WPtrs wp, ushort_t* __restrict__ wx_t, ushort_t* __restrict__ wh_t) {
    long idx = (long)blockIdx.x * 256 + threadIdx.x;        // over 6144*512
    int n = (int)(idx >> 9);
    int k = (int)(idx & 511);
    if (n < NX) {
        int g = n >> 9, h = n & 511;
        wx_t[(long)n * 512 + k] = f2bf(wp.p[g][(long)k * HH + h]);
    } else {
        int n2 = n - NX;
        int g = n2 >> 9, h = n2 & 511;
        wh_t[(long)n2 * 512 + k] = f2bf(wp.p[8 + g][(long)k * HH + h]);
    }
}

__global__ void init_state_kernel(ushort_t* __restrict__ h0, float* __restrict__ c0) {
    int idx = blockIdx.x * 256 + threadIdx.x;               // over 256*512
    h0[idx] = 0;
    c0[idx] = 0.0f;
}

// ---------------------------------------------------------------------------
// Phase 1: X @ Wx_all   M=65536 N=4096 K=512, block tile 128x128, 8 waves
// wave w: M-sub 32*(w&3), N-sub 64*(w>>2)  -> 2x4 grid of 16x16x32 WMMAs
// K-chunks of 32 staged in LDS via async copies, double buffered.
// LDS tiles padded to 40 ushorts/row (80B) to spread DS bank groups.
// ---------------------------------------------------------------------------
#define TPAD   40                 // padded row pitch (ushorts) for a 32-wide K chunk
#define TILE_U (128 * TPAD)       // one 128x32 tile in ushorts (5120)

struct GateParams {
    const float* bias[8];                 // bi bf bc bt1 bt2 bd1 bd2 bo
    const float* wt1; const float* wt2;
    const float* wd1; const float* wd2;
    const float* wto; const float* wdo;
};

__global__ __launch_bounds__(256, 1)
void gemm_x_kernel(const ushort_t* __restrict__ X, const ushort_t* __restrict__ Wt,
                   float* __restrict__ gates, const float* __restrict__ TT,
                   const float* __restrict__ DT, GateParams gp) {
    __shared__ ushort_t smem[4 * TILE_U];   // A[2 bufs], B[2 bufs] : 40 KB

    const int wave = threadIdx.x >> 5;
    const int lane = threadIdx.x & 31;
    const int r    = lane & 15;
    const int hi   = lane >> 4;

    const int mb0 = blockIdx.y * 128;
    const int nb0 = blockIdx.x * 128;
    const int msub = (wave & 3) * 32;
    const int nsub = (wave >> 2) * 64;
    const int g = (int)(blockIdx.x >> 2);                   // gate is uniform per block

    const unsigned smem_base = (unsigned)(uintptr_t)(&smem[0]);

    floatx8 acc[2][4] = {};

    // issue async copies of one 128x32 A tile + 128x32 B tile into buffer p
    auto issue_chunk = [&](int p, int kc) {
        unsigned abase = smem_base + (unsigned)(p * TILE_U) * 2u;
        unsigned bbase = smem_base + (unsigned)((2 + p) * TILE_U) * 2u;
#pragma unroll
        for (int c = (int)threadIdx.x; c < 512; c += 256) {
            int row = c >> 2, q = c & 3;
            async_copy_b128(abase + (unsigned)(row * TPAD + q * 8) * 2u,
                            X + (long)(mb0 + row) * 512 + kc + q * 8);
            async_copy_b128(bbase + (unsigned)(row * TPAD + q * 8) * 2u,
                            Wt + (long)(nb0 + row) * 512 + kc + q * 8);
        }
    };

    issue_chunk(0, 0);
    wait_async0();
    __syncthreads();

    for (int it = 0; it < 16; ++it) {
        if (it + 1 < 16) issue_chunk((it + 1) & 1, (it + 1) * 32);

        const ushort_t* As = smem + (it & 1) * TILE_U;
        const ushort_t* Bs = smem + (2 + (it & 1)) * TILE_U;

        Frag a[2];
#pragma unroll
        for (int i = 0; i < 2; ++i) {
            const ushort_t* ap = As + (msub + 16 * i + r) * TPAD + hi * 8;
            a[i].q[0] = *(const uint4*)(ap);
            a[i].q[1] = *(const uint4*)(ap + 16);
        }
        Frag b[4];
#pragma unroll
        for (int j = 0; j < 4; ++j) {
            const ushort_t* bp = Bs + (nsub + 16 * j + r) * TPAD + hi * 8;
            b[j].q[0] = *(const uint4*)(bp);
            b[j].q[1] = *(const uint4*)(bp + 16);
        }
#pragma unroll
        for (int i = 0; i < 2; ++i)
#pragma unroll
            for (int j = 0; j < 4; ++j)
                acc[i][j] = __builtin_amdgcn_wmma_f32_16x16x32_bf16(
                    false, a[i].v, false, b[j].v, (short)0, acc[i][j], false, false);

        if (it + 1 < 16) {
            wait_async0();          // next chunk landed in LDS
            __syncthreads();        // all waves done reading current buffers
        }
    }

    // epilogue: add bias / time-distance gate terms, write [s][b][gate][h] fp32
    const int col = lane & 15;
#pragma unroll
    for (int i = 0; i < 2; ++i) {
#pragma unroll
        for (int j = 0; j < 4; ++j) {
#pragma unroll
            for (int v = 0; v < 8; ++v) {
                int m = mb0 + msub + 16 * i + v + 8 * hi;
                int n = nb0 + nsub + 16 * j + col;
                int bidx = m >> 8;          // batch
                int s    = m & 255;         // timestep
                int h    = n & 511;
                float add;
                switch (g) {
                    case 0: add = gp.bias[0][h]; break;
                    case 1: add = gp.bias[1][h]; break;
                    case 2: add = gp.bias[2][h]; break;
                    case 3: add = sigf(TT[m] * gp.wt1[h]) + gp.bias[3][h]; break;
                    case 4: add = sigf(TT[m] * gp.wt2[h]) + gp.bias[4][h]; break;
                    case 5: add = sigf(DT[m] * gp.wd1[h]) + gp.bias[5][h]; break;
                    case 6: add = sigf(DT[m] * gp.wd2[h]) + gp.bias[6][h]; break;
                    default: add = TT[m] * gp.wto[h] + DT[m] * gp.wdo[h] + gp.bias[7][h]; break;
                }
                gates[((long)(s * BB + bidx) * 8 + g) * 512 + h] = acc[i][j][v] + add;
            }
        }
    }
}

// ---------------------------------------------------------------------------
// Phase 2: one launch per timestep. Block = 128 threads (4 waves = 4 gates).
// Each block owns a 32(b) x 32(h) tile; wave g computes h_prev @ Wh_g for it,
// tiles exchanged through LDS, then fused pointwise STGN cell update.
// ---------------------------------------------------------------------------
__global__ __launch_bounds__(128, 1)
void step_kernel(const ushort_t* __restrict__ Hin, ushort_t* __restrict__ Hout,
                 const ushort_t* __restrict__ WhT, const float* __restrict__ gates,
                 float* __restrict__ Cbuf, float* __restrict__ hidden,
                 float* __restrict__ out_ht, float* __restrict__ out_ct, int t) {
    const int wave = threadIdx.x >> 5;    // gate: 0=i 1=f 2=c 3=o
    const int lane = threadIdx.x & 31;
    const int r    = lane & 15;
    const int hi   = lane >> 4;

    const int m0 = blockIdx.y * 32;       // batch rows
    const int h0 = blockIdx.x * 32;       // hidden cols

    floatx8 acc[2][2] = {};

    for (int kc = 0; kc < 512; kc += 32) {
        Frag a[2];
#pragma unroll
        for (int i = 0; i < 2; ++i) {
            const ushort_t* ap = Hin + (long)(m0 + 16 * i + r) * 512 + kc + hi * 8;
            a[i].q[0] = *(const uint4*)(ap);
            a[i].q[1] = *(const uint4*)(ap + 16);
            if (kc + 32 < 512) __builtin_prefetch(ap + 32, 0, 0);
        }
        Frag b[2];
#pragma unroll
        for (int j = 0; j < 2; ++j) {
            const ushort_t* bp = WhT + (long)(wave * 512 + h0 + 16 * j + r) * 512 + kc + hi * 8;
            b[j].q[0] = *(const uint4*)(bp);
            b[j].q[1] = *(const uint4*)(bp + 16);
        }
#pragma unroll
        for (int i = 0; i < 2; ++i)
#pragma unroll
            for (int j = 0; j < 2; ++j)
                acc[i][j] = __builtin_amdgcn_wmma_f32_16x16x32_bf16(
                    false, a[i].v, false, b[j].v, (short)0, acc[i][j], false, false);
    }

    __shared__ float lds[4][32][33];
    const int col = lane & 15;
#pragma unroll
    for (int i = 0; i < 2; ++i)
#pragma unroll
        for (int j = 0; j < 2; ++j)
#pragma unroll
            for (int v = 0; v < 8; ++v)
                lds[wave][16 * i + v + 8 * hi][16 * j + col] = acc[i][j][v];
    __syncthreads();

    // pointwise STGN cell update: 32x32 = 1024 elements over 128 threads
    for (int e = threadIdx.x; e < 1024; e += 128) {
        int bl = e >> 5, hl = e & 31;
        int b = m0 + bl, h = h0 + hl;
        long base = ((long)(t * BB + b) * 8) * 512 + h;
        float xi = gates[base + 0 * 512] + lds[0][bl][hl];
        float xf = gates[base + 1 * 512] + lds[1][bl][hl];
        float xc = gates[base + 2 * 512] + lds[2][bl][hl];
        float t1 = gates[base + 3 * 512];
        float t2 = gates[base + 4 * 512];
        float d1 = gates[base + 5 * 512];
        float d2 = gates[base + 6 * 512];
        float po = gates[base + 7 * 512] + lds[3][bl][hl];

        float it = sigf(xi), ft = sigf(xf), jt = tanhf(xc);
        float T1 = sigf(t1), T2 = sigf(t2), D1 = sigf(d1), D2 = sigf(d2);
        float c  = Cbuf[b * 512 + h];
        float fc = ft * c;
        float ij = it * jt;
        float c_hat = fc + ij * (T1 * D1);
        float c_new = fc + ij * (T2 * D2);
        float ot = sigf(po);
        float h_new = ot * tanhf(c_hat);

        Cbuf[b * 512 + h] = c_new;
        Hout[b * 512 + h] = f2bf(h_new);
        hidden[((long)b * SS + t) * HH + h] = h_new;
        if (t == SS - 1) {
            out_ht[b * 512 + h] = h_new;
            out_ct[b * 512 + h] = c_new;
        }
    }
}

// ---------------------------------------------------------------------------
// Launcher
// ---------------------------------------------------------------------------
extern "C" void kernel_launch(void* const* d_in, const int* in_sizes, int n_in,
                              void* d_out, int out_size, void* d_ws, size_t ws_size,
                              hipStream_t stream) {
    const float* lstm_input = (const float*)d_in[0];

    // workspace layout
    char* ws = (char*)d_ws;
    size_t off = 0;
    auto alloc = [&](size_t bytes) { char* p = ws + off; off = (off + bytes + 255) & ~(size_t)255; return p; };
    ushort_t* Xbf   = (ushort_t*)alloc((size_t)MROWS * 512 * 2);
    float*    TT    = (float*)   alloc((size_t)MROWS * 4);
    float*    DT    = (float*)   alloc((size_t)MROWS * 4);
    ushort_t* WxT   = (ushort_t*)alloc((size_t)NX * 512 * 2);
    ushort_t* WhT   = (ushort_t*)alloc((size_t)NR * 512 * 2);
    float*    gates = (float*)   alloc((size_t)MROWS * NX * 4);
    ushort_t* Hbuf0 = (ushort_t*)alloc((size_t)BB * HH * 2);
    ushort_t* Hbuf1 = (ushort_t*)alloc((size_t)BB * HH * 2);
    float*    Cbuf  = (float*)   alloc((size_t)BB * HH * 4);

    float* out_hidden = (float*)d_out;                       // (B,S,H)
    float* out_ht     = out_hidden + (long)BB * SS * HH;     // (B,H)
    float* out_ct     = out_ht + (long)BB * HH;              // (B,H)

    // pack inputs -> bf16
    pack_x_kernel<<<(MROWS * 512) / 256, 256, 0, stream>>>(lstm_input, Xbf, TT, DT);

    WPtrs wp;
    wp.p[0] = (const float*)d_in[1];   // Wxi
    wp.p[1] = (const float*)d_in[4];   // Wxf
    wp.p[2] = (const float*)d_in[7];   // Wxc
    wp.p[3] = (const float*)d_in[10];  // Wxt1
    wp.p[4] = (const float*)d_in[13];  // Wxt2
    wp.p[5] = (const float*)d_in[16];  // Wxd1
    wp.p[6] = (const float*)d_in[19];  // Wxd2
    wp.p[7] = (const float*)d_in[22];  // Wxo
    wp.p[8]  = (const float*)d_in[2];  // Whi
    wp.p[9]  = (const float*)d_in[5];  // Whf
    wp.p[10] = (const float*)d_in[8];  // Whc
    wp.p[11] = (const float*)d_in[23]; // Who
    pack_w_kernel<<<((NX + NR) * 512) / 256, 256, 0, stream>>>(wp, WxT, WhT);

    init_state_kernel<<<(BB * HH) / 256, 256, 0, stream>>>(Hbuf0, Cbuf);

    GateParams gp;
    gp.bias[0] = (const float*)d_in[3];   // bi
    gp.bias[1] = (const float*)d_in[6];   // bf
    gp.bias[2] = (const float*)d_in[9];   // bc
    gp.bias[3] = (const float*)d_in[12];  // bt1
    gp.bias[4] = (const float*)d_in[15];  // bt2
    gp.bias[5] = (const float*)d_in[18];  // bd1
    gp.bias[6] = (const float*)d_in[21];  // bd2
    gp.bias[7] = (const float*)d_in[26];  // bo
    gp.wt1 = (const float*)d_in[11];
    gp.wt2 = (const float*)d_in[14];
    gp.wd1 = (const float*)d_in[17];
    gp.wd2 = (const float*)d_in[20];
    gp.wto = (const float*)d_in[24];
    gp.wdo = (const float*)d_in[25];

    // Phase 1: big input-projection GEMM (M=65536, N=4096, K=512)
    gemm_x_kernel<<<dim3(NX / 128, MROWS / 128), 256, 0, stream>>>(Xbf, WxT, gates, TT, DT, gp);

    // Phase 2: sequential scan, ping-pong h buffers
    for (int t = 0; t < SS; ++t) {
        ushort_t* hin  = (t & 1) ? Hbuf1 : Hbuf0;
        ushort_t* hout = (t & 1) ? Hbuf0 : Hbuf1;
        step_kernel<<<dim3(HH / 32, BB / 32), 128, 0, stream>>>(
            hin, hout, WhT, gates, Cbuf, out_hidden, out_ht, out_ct, t);
    }
    (void)in_sizes; (void)n_in; (void)out_size; (void)ws_size;
}
